// LinearCombineVC_5368709120212
// MI455X (gfx1250) — matrix operand order repaired and checked by reading, not compiled, not run on hardware
//
#include <hip/hip_runtime.h>
#include <hip/hip_bf16.h>
#include <cstdint>

// ---------------------------------------------------------------------------
// Problem dims fixed by reference setup_inputs()
// ---------------------------------------------------------------------------
constexpr int NB = 16;            // batch
constexpr int C  = 256;           // channels (K of GEMM1, M of GEMM2)
constexpr int L  = 32 * 32;       // pixels  (M of GEMM1, N of GEMM2)
constexpr int P  = 2048;          // concept pool (N of GEMM1, K of GEMM2)
constexpr int KEEP = (int)(0.05 * P);   // 102, matches int(topk * P)

typedef __attribute__((ext_vector_type(16))) __bf16 v16bf;
typedef __attribute__((ext_vector_type(8)))  __bf16 v8bf;
typedef __attribute__((ext_vector_type(8)))  float  v8f;
typedef __attribute__((ext_vector_type(4)))  float  v4f;

static __device__ __forceinline__ v16bf combine16(v8bf lo, v8bf hi) {
    v16bf r;
#pragma unroll
    for (int i = 0; i < 8; ++i) { r[i] = lo[i]; r[i + 8] = hi[i]; }
    return r;
}

// ---------------------------------------------------------------------------
// pool [P][C] f32  ->  pool_bf16 [P][C]  +  inv_pool_norm [P]
// one wave32 per pool row
// ---------------------------------------------------------------------------
__global__ __launch_bounds__(256) void pool_prep(const float* __restrict__ pool,
                                                 __bf16* __restrict__ pool_bf,
                                                 float* __restrict__ inv_pn) {
    int tid  = threadIdx.x;
    int lane = tid & 31;
    int p    = blockIdx.x * 8 + (tid >> 5);
    const float* row = pool + (size_t)p * C;
    __bf16* orow     = pool_bf + (size_t)p * C;
    float s = 0.f;
    for (int c = lane; c < C; c += 32) {
        float v = row[c];
        orow[c] = (__bf16)v;
        s += v * v;
    }
#pragma unroll
    for (int off = 16; off > 0; off >>= 1) s += __shfl_down(s, off, 32);
    if (lane == 0) inv_pn[p] = rsqrtf(s);
}

// ---------------------------------------------------------------------------
// Tiled transpose + f32->bf16 convert:  src[z][R][Cc] -> dst[z][Cc][R]
// ---------------------------------------------------------------------------
__global__ __launch_bounds__(256) void transpose_f32_to_bf16(
        const float* __restrict__ src, __bf16* __restrict__ dst, int R, int Cc) {
    __shared__ float tile[32][33];
    int tx = threadIdx.x & 31;
    int ty = threadIdx.x >> 5;                 // 0..7
    int r0 = blockIdx.x * 32;
    int c0 = blockIdx.y * 32;
    const float* s = src + (size_t)blockIdx.z * R * Cc;
    __bf16*      d = dst + (size_t)blockIdx.z * R * Cc;
#pragma unroll
    for (int i = 0; i < 4; ++i) {
        int rr = ty * 4 + i;
        tile[rr][tx] = s[(size_t)(r0 + rr) * Cc + c0 + tx];
    }
    __syncthreads();
#pragma unroll
    for (int i = 0; i < 4; ++i) {
        int cc = ty * 4 + i;
        d[(size_t)(c0 + cc) * R + r0 + tx] = (__bf16)tile[tx][cc];
    }
}

// ---------------------------------------------------------------------------
// inv_xn[n*L + l] = rsqrt( sum_c x[n][c][l]^2 )    (coalesced along l)
// ---------------------------------------------------------------------------
__global__ __launch_bounds__(256) void x_inv_norm(const float* __restrict__ x,
                                                  float* __restrict__ inv_xn) {
    int idx = blockIdx.x * blockDim.x + threadIdx.x;   // n*L + l
    int n = idx / L, l = idx - n * L;
    const float* base = x + (size_t)n * C * L + l;
    float s = 0.f;
#pragma unroll 4
    for (int c = 0; c < C; ++c) { float v = base[(size_t)c * L]; s += v * v; }
    inv_xn[idx] = rsqrtf(s);
}

// ---------------------------------------------------------------------------
// bf16 WMMA GEMM:  D[z][M][N] (+scales) = A[z][M][K] * Bt[z][N][K]^T
//   A row-major   -> A-fragment, lane half h holds k 16h..16h+15 (2x16B contiguous)
//   Bt row-major  -> B-fragment, lane half h holds k 16h..16h+15 (2x16B contiguous)
// 256 threads = 8 wave32; wave tile 64x64 (4x4 fragments, 16 WMMA / K-step);
// block tile 128x256. Prefetches are unconditional: default-TH prefetch is
// speculative and silently dropped on invalid translation, so running a few
// cachelines past the buffer end on the last K-step is free of side effects
// and keeps the inner loop branch-free.
// ---------------------------------------------------------------------------
__global__ __launch_bounds__(256) void gemm_bf16_wmma(
        const __bf16* __restrict__ A, const __bf16* __restrict__ Bt,
        float* __restrict__ D, int M, int N, int K,
        size_t strA, size_t strB, size_t strD,
        const float* __restrict__ rscale, size_t strR,
        const float* __restrict__ cscale) {
    int z = blockIdx.z;
    A  += (size_t)z * strA;
    Bt += (size_t)z * strB;
    D  += (size_t)z * strD;
    const float* rs = rscale ? rscale + (size_t)z * strR : nullptr;

    int tid  = threadIdx.x;
    int lane = tid & 31;
    int wave = tid >> 5;
    int wm   = wave >> 2;                       // 0..1
    int wn   = wave & 3;                        // 0..3
    int m_base = blockIdx.y * 128 + wm * 64;
    int n_base = blockIdx.x * 256 + wn * 64;

    int l15    = lane & 15;
    int hiHalf = lane >> 4;                     // 0 | 1
    int aoff   = hiHalf * 16;                   // A-fragment k sub-offset
    int boff   = hiHalf * 16;                   // B-fragment k sub-offset

    v8f acc[4][4];
#pragma unroll
    for (int i = 0; i < 4; ++i)
#pragma unroll
        for (int j = 0; j < 4; ++j)
#pragma unroll
            for (int v = 0; v < 8; ++v) acc[i][j][v] = 0.f;

    for (int k0 = 0; k0 < K; k0 += 32) {
        v16bf a[4], b[4];
#pragma unroll
        for (int i = 0; i < 4; ++i) {
            const __bf16* pa = A + (size_t)(m_base + 16 * i + l15) * K + k0 + aoff;
            v8bf lo = *(const v8bf*)pa;
            v8bf hi = *(const v8bf*)(pa + 8);
            a[i] = combine16(lo, hi);
            __builtin_prefetch(pa + 32, 0, 3);   // speculative global_prefetch_b8
        }
#pragma unroll
        for (int j = 0; j < 4; ++j) {
            const __bf16* pb = Bt + (size_t)(n_base + 16 * j + l15) * K + k0 + boff;
            v8bf lo = *(const v8bf*)pb;
            v8bf hi = *(const v8bf*)(pb + 8);
            b[j] = combine16(lo, hi);
            __builtin_prefetch(pb + 32, 0, 3);   // speculative global_prefetch_b8
        }
#pragma unroll
        for (int i = 0; i < 4; ++i)
#pragma unroll
            for (int j = 0; j < 4; ++j)
                acc[i][j] = __builtin_amdgcn_wmma_f32_16x16x32_bf16(
                        false, a[i], false, b[j], (short)0, acc[i][j], false, false);
    }

    // C/D layout: lanes 0-15 -> N=lane, M=v ; lanes 16-31 -> N=lane-16, M=v+8
#pragma unroll
    for (int j = 0; j < 4; ++j) {
        int col  = n_base + 16 * j + l15;
        float cs = cscale ? cscale[col] : 1.f;
#pragma unroll
        for (int i = 0; i < 4; ++i) {
            int rbase = m_base + 16 * i + 8 * hiHalf;
            float* dp = D + (size_t)rbase * N + col;
#pragma unroll
            for (int v = 0; v < 8; ++v) {
                float rv = rs ? rs[rbase + v] : 1.f;
                dp[(size_t)v * N] = acc[i][j][v] * cs * rv;
            }
        }
    }
}

// ---------------------------------------------------------------------------
// Exact top-KEEP select over P=2048 |cos| values per (n,l) row, jax-style
// index-order tie break, renormalize over the kept entries, emit bf16 w[l][p].
// One 256-thread block per row; 8 values/thread kept in registers.
// ---------------------------------------------------------------------------
__global__ __launch_bounds__(256) void topk_normalize(const float* __restrict__ cosd,
                                                      __bf16* __restrict__ wout) {
    __shared__ int   s_part[8];
    __shared__ int   s_int;
    __shared__ float s_fpart[8];
    __shared__ float s_flt;
    __shared__ int   s_scan[256];

    int tid  = threadIdx.x;
    int lane = tid & 31;
    int warp = tid >> 5;
    size_t row = blockIdx.x;
    const float* base = cosd + row * (size_t)P;

    float    vals[8];
    uint32_t keys[8];
    {
        v4f a = *(const v4f*)(base + tid * 8);
        v4f b = *(const v4f*)(base + tid * 8 + 4);
#pragma unroll
        for (int i = 0; i < 4; ++i) { vals[i] = a[i]; vals[4 + i] = b[i]; }
    }
#pragma unroll
    for (int i = 0; i < 8; ++i) keys[i] = __float_as_uint(fabsf(vals[i]));

    auto blockSumI = [&](int v) -> int {
#pragma unroll
        for (int off = 16; off > 0; off >>= 1) v += __shfl_down(v, off, 32);
        if (lane == 0) s_part[warp] = v;
        __syncthreads();
        if (tid == 0) { int t = 0; for (int i = 0; i < 8; ++i) t += s_part[i]; s_int = t; }
        __syncthreads();
        return s_int;
    };
    auto blockSumF = [&](float v) -> float {
#pragma unroll
        for (int off = 16; off > 0; off >>= 1) v += __shfl_down(v, off, 32);
        if (lane == 0) s_fpart[warp] = v;
        __syncthreads();
        if (tid == 0) { float t = 0.f; for (int i = 0; i < 8; ++i) t += s_fpart[i]; s_flt = t; }
        __syncthreads();
        return s_flt;
    };

    // KEEP-th largest key: max T with count(key >= T) >= KEEP (|x| bits monotonic)
    uint32_t loT = 0, hiT = 0xFFFFFFFFu;
    while (loT < hiT) {
        uint32_t mid = loT + ((hiT - loT + 1) >> 1);
        int c = 0;
#pragma unroll
        for (int i = 0; i < 8; ++i) c += (keys[i] >= mid);
        if (blockSumI(c) >= KEEP) loT = mid; else hiT = mid - 1;
        __syncthreads();
    }
    uint32_t thr = loT;

    int g = 0, e = 0;
#pragma unroll
    for (int i = 0; i < 8; ++i) { g += (keys[i] > thr); e += (keys[i] == thr); }
    int cntG = blockSumI(g);
    int rem  = KEEP - cntG;                    // ties kept in index order

    // exclusive scan of per-thread tie counts (index order == thread order)
    s_scan[tid] = e;
    __syncthreads();
    for (int off = 1; off < 256; off <<= 1) {
        int t = (tid >= off) ? s_scan[tid - off] : 0;
        __syncthreads();
        s_scan[tid] += t;
        __syncthreads();
    }
    int excl = s_scan[tid] - e;

    bool  sel[8];
    float ssum = 0.f;
    int eq_seen = 0;
#pragma unroll
    for (int i = 0; i < 8; ++i) {
        bool s;
        if (keys[i] > thr)        s = true;
        else if (keys[i] == thr)  { s = (excl + eq_seen) < rem; ++eq_seen; }
        else                      s = false;
        sel[i] = s;
        if (s) ssum += vals[i];
    }
    float inv = 1.0f / blockSumF(ssum);        // normed = masked / sum(masked)

    v8bf o;
#pragma unroll
    for (int i = 0; i < 8; ++i) o[i] = sel[i] ? (__bf16)(vals[i] * inv) : (__bf16)0.f;
    *(v8bf*)(wout + row * (size_t)P + tid * 8) = o;
}

// ---------------------------------------------------------------------------
extern "C" void kernel_launch(void* const* d_in, const int* in_sizes, int n_in,
                              void* d_out, int out_size, void* d_ws, size_t ws_size,
                              hipStream_t stream) {
    const float* x    = (const float*)d_in[0];   // [NB][C][L]
    const float* pool = (const float*)d_in[1];   // [P][C]
    float* out = (float*)d_out;                  // [NB][C][L]

    char* ws = (char*)d_ws;
    size_t off = 0;
    auto alloc = [&](size_t bytes) -> void* {
        void* p = ws + off;
        off += (bytes + 255) & ~(size_t)255;
        return p;
    };
    float*  cosb    = (float*) alloc((size_t)NB * L * P * 4);  // cos_t [n][l][p]
    __bf16* wgt     = (__bf16*)alloc((size_t)NB * L * P * 2);  // w     [n][l][p]
    __bf16* xT      = (__bf16*)alloc((size_t)NB * L * C * 2);  // A1    [n][l][c]
    __bf16* pool_bf = (__bf16*)alloc((size_t)P * C * 2);       // Bt1   [p][c]
    __bf16* poolT   = (__bf16*)alloc((size_t)C * P * 2);       // A2    [c][p]
    float*  inv_pn  = (float*) alloc((size_t)P * 4);
    float*  inv_xn  = (float*) alloc((size_t)NB * L * 4);
    (void)ws_size; (void)in_sizes; (void)n_in; (void)out_size;

    pool_prep<<<P / 8, 256, 0, stream>>>(pool, pool_bf, inv_pn);
    transpose_f32_to_bf16<<<dim3(P / 32, C / 32, 1), 256, 0, stream>>>(pool, poolT, P, C);
    x_inv_norm<<<(NB * L) / 256, 256, 0, stream>>>(x, inv_xn);
    transpose_f32_to_bf16<<<dim3(C / 32, L / 32, NB), 256, 0, stream>>>(x, xT, C, L);

    // GEMM1: cos[n][l][p] = (xT * pool^T) scaled by rsqrt(|x|^2) * rsqrt(|pool|^2)
    gemm_bf16_wmma<<<dim3(P / 256, L / 128, NB), 256, 0, stream>>>(
            xT, pool_bf, cosb, L, P, C,
            (size_t)L * C, 0, (size_t)L * P,
            inv_xn, (size_t)L, inv_pn);

    topk_normalize<<<NB * L, 256, 0, stream>>>(cosb, wgt);

    // GEMM2: out[n][c][l] = poolT * w^T   (weights already normalized)
    gemm_bf16_wmma<<<dim3(L / 256, C / 128, NB), 256, 0, stream>>>(
            poolT, wgt, out, C, L, P,
            0, (size_t)L * P, (size_t)C * L,
            nullptr, 0, nullptr);
}
